// LSTM_GCN_Baseline_79370995630959
// MI455X (gfx1250) — matrix operand order repaired and checked by reading, not compiled, not run on hardware
//
#include <hip/hip_runtime.h>
#include <hip/hip_bf16.h>

typedef __attribute__((ext_vector_type(16))) _Float16 v16h;
typedef __attribute__((ext_vector_type(8)))  float    v8f;

constexpr int Bc = 8, Tc = 24, Nc = 4000, Fc = 32, Hc = 128, Ec = 64000;
constexpr int G4 = 4 * Hc;   // 512 gate columns
constexpr int BN = Bc * Nc;  // 32000 sequences / nodes total

__device__ __forceinline__ float sigf(float x) { return 1.0f / (1.0f + __expf(-x)); }
__device__ __forceinline__ float tanhfast(float x) {
  const float xc = fminf(fmaxf(x, -15.0f), 15.0f);
  const float e  = __expf(2.0f * xc);
  return (e - 1.0f) / (e + 1.0f);
}

// ---------------------------------------------------------------------------
// LSTM: 32 rows per block, 8 waves; wave w owns gate cols [64w, 64w+64).
// W_ih^T / W_hh^T live in VGPRs as WMMA B-fragments for the whole recurrence.
// h, c stay on-chip across all 24 steps; x tiles are software-pipelined
// (global->regs during WMMA phase, regs->LDS during elementwise phase).
// 40 x v_wmma_f32_16x16x32_f16 per wave per step, fp32 accumulation.
// ---------------------------------------------------------------------------
__global__ __launch_bounds__(256) void lstm_kernel(
    const float* __restrict__ dyn,   // [B,T,N,F]
    const float* __restrict__ W_ih,  // [4H,F]
    const float* __restrict__ W_hh,  // [4H,H]
    const float* __restrict__ b_ih,  // [4H]
    const float* __restrict__ b_hh,  // [4H]
    float* __restrict__ h_out)       // [B*N,H]
{
  __shared__ _Float16 x16[2][32][Fc];   // double-buffered staged x tile (f16)
  __shared__ _Float16 h16s[32][Hc];     // hidden state (f16) for A-frags
  __shared__ float    gpre[32][G4];     // gate pre-activations (f32)

  const int tid  = threadIdx.x;
  const int wv   = tid >> 5;
  const int lane = tid & 31;
  const int ln   = lane & 15;
  const int m0   = blockIdx.x * 32;
  const int bb   = m0 / Nc;             // batch (32-row blocks never straddle batches: N%32==0)
  const int n0   = m0 - bb * Nc;

  // ---- preload B fragments (contiguous K order, same as A: lane<16 -> K=kb..kb+15, lane>=16 -> K=kb+16..kb+31)
  const int kb16 = (lane < 16) ? 0 : 16;
  v16h Bih[4];
  v16h Bhh[4][4];
  float bias[4];
#pragma unroll
  for (int ct = 0; ct < 4; ++ct) {
    const int col = wv * 64 + ct * 16 + ln;      // gate column in [0,512)
    bias[ct] = b_ih[col] + b_hh[col];
#pragma unroll
    for (int h = 0; h < 16; ++h)
      Bih[ct][h] = (_Float16)W_ih[col * Fc + kb16 + h];
#pragma unroll
    for (int kf = 0; kf < 4; ++kf)
#pragma unroll
      for (int h = 0; h < 16; ++h)
        Bhh[ct][kf][h] = (_Float16)W_hh[col * Hc + kf * 32 + kb16 + h];
  }

  // ---- zero-init h (LDS) and c (registers, 16 elems/thread of the 32x128 tile)
  float creg[16];
#pragma unroll
  for (int k = 0; k < 16; ++k) creg[k] = 0.0f;
  for (int e = tid; e < 32 * Hc; e += 256) ((_Float16*)h16s)[e] = (_Float16)0.0f;

  // ---- prologue: stage x(t=0) into buffer 0
  for (int e = tid; e < 32 * Fc; e += 256) {
    const int r = e >> 5, f = e & 31;
    x16[0][r][f] = (_Float16)dyn[(((long)bb * Tc + 0) * Nc + (n0 + r)) * Fc + f];
  }
  __syncthreads();

  // 16-bit A layout (matches B): lane<16 -> K0-15, lane>=16 -> K16-31
  const int kA0 = (lane < 16) ? 0 : 16;
  const int rb  = (lane < 16) ? 0 : 8;

  for (int t = 0; t < Tc; ++t) {
    const int cur = t & 1, nxt = cur ^ 1;

    // software pipeline: pull x(t+1) into registers while WMMAs run
    float xpre[4];
    if (t + 1 < Tc) {
#pragma unroll
      for (int q = 0; q < 4; ++q) {
        const int e = tid + 256 * q;
        const int r = e >> 5, f = e & 31;
        xpre[q] = dyn[(((long)bb * Tc + (t + 1)) * Nc + (n0 + r)) * Fc + f];
      }
    }

    // A fragments from LDS (two 16-row tiles)
    v16h Ax[2];
    v16h Ah[2][4];
#pragma unroll
    for (int rt = 0; rt < 2; ++rt) {
      const int row = rt * 16 + ln;
#pragma unroll
      for (int h = 0; h < 8; ++h) {
        Ax[rt][h]     = x16[cur][row][kA0 + h];
        Ax[rt][8 + h] = x16[cur][row][kA0 + 8 + h];
      }
#pragma unroll
      for (int kf = 0; kf < 4; ++kf)
#pragma unroll
        for (int h = 0; h < 8; ++h) {
          Ah[rt][kf][h]     = h16s[row][kf * 32 + kA0 + h];
          Ah[rt][kf][8 + h] = h16s[row][kf * 32 + kA0 + 8 + h];
        }
    }

    // 4 col tiles x 2 row tiles x (1 + 4) WMMAs, fp32 accumulation seeded with bias
#pragma unroll
    for (int ct = 0; ct < 4; ++ct) {
      const int ccol = wv * 64 + ct * 16 + ln;
#pragma unroll
      for (int rt = 0; rt < 2; ++rt) {
        v8f acc;
#pragma unroll
        for (int i = 0; i < 8; ++i) acc[i] = bias[ct];
        acc = __builtin_amdgcn_wmma_f32_16x16x32_f16(false, Ax[rt], false, Bih[ct],
                                                     (short)0, acc, false, false);
#pragma unroll
        for (int kf = 0; kf < 4; ++kf)
          acc = __builtin_amdgcn_wmma_f32_16x16x32_f16(false, Ah[rt][kf], false, Bhh[ct][kf],
                                                       (short)0, acc, false, false);
#pragma unroll
        for (int v = 0; v < 8; ++v) gpre[rt * 16 + rb + v][ccol] = acc[v];  // C layout -> LDS
      }
    }
    __syncthreads();

    // fused cell update: i,f,g,o aligned per hidden column; c in registers
#pragma unroll
    for (int k = 0; k < 16; ++k) {
      const int idx = tid + 256 * k;              // 0..4095 over 32x128
      const int r = idx >> 7, j = idx & 127;
      const float gi = gpre[r][j];
      const float gf = gpre[r][Hc + j];
      const float gg = gpre[r][2 * Hc + j];
      const float go = gpre[r][3 * Hc + j];
      const float cn = sigf(gf) * creg[k] + sigf(gi) * tanhfast(gg);
      const float hn = sigf(go) * tanhfast(cn);
      creg[k] = cn;
      h16s[r][j] = (_Float16)hn;
      if (t == Tc - 1) h_out[(long)(m0 + r) * Hc + j] = hn;
    }
    // commit prefetched x(t+1) to the alternate LDS buffer
    if (t + 1 < Tc) {
#pragma unroll
      for (int q = 0; q < 4; ++q) {
        const int e = tid + 256 * q;
        const int r = e >> 5, f = e & 31;
        x16[nxt][r][f] = (_Float16)xpre[q];
      }
    }
    __syncthreads();
  }
}

// ---------------------------------------------------------------------------
// xw = h @ gcn_W^T : 16 rows/block, 8 waves, wave w -> output cols [16w,16w+16)
// ---------------------------------------------------------------------------
__global__ __launch_bounds__(256) void xw_kernel(
    const float* __restrict__ h_in, const float* __restrict__ gcn_W,
    float* __restrict__ xw)
{
  __shared__ _Float16 a16[16][Hc];
  const int tid = threadIdx.x, wv = tid >> 5, lane = tid & 31, ln = lane & 15;
  const int m0 = blockIdx.x * 16;
  const int kb16 = (lane < 16) ? 0 : 16;
  const int kA0  = (lane < 16) ? 0 : 16;

  v16h Bg[4];
  const int col = wv * 16 + ln;
#pragma unroll
  for (int kf = 0; kf < 4; ++kf)
#pragma unroll
    for (int h = 0; h < 16; ++h)
      Bg[kf][h] = (_Float16)gcn_W[col * Hc + kf * 32 + kb16 + h];

  for (int e = tid; e < 16 * Hc; e += 256) {
    const int r = e >> 7, k = e & 127;
    a16[r][k] = (_Float16)h_in[(long)(m0 + r) * Hc + k];
  }
  __syncthreads();

  v16h Ah[4];
#pragma unroll
  for (int kf = 0; kf < 4; ++kf)
#pragma unroll
    for (int h = 0; h < 8; ++h) {
      Ah[kf][h]     = a16[ln][kf * 32 + kA0 + h];
      Ah[kf][8 + h] = a16[ln][kf * 32 + kA0 + 8 + h];
    }

  v8f acc = {};
#pragma unroll
  for (int kf = 0; kf < 4; ++kf)
    acc = __builtin_amdgcn_wmma_f32_16x16x32_f16(false, Ah[kf], false, Bg[kf],
                                                 (short)0, acc, false, false);
  const int rb = (lane < 16) ? 0 : 8;
#pragma unroll
  for (int v = 0; v < 8; ++v)
    xw[(long)(m0 + rb + v) * Hc + col] = acc[v];
}

// ---------------------------------------------------------------------------
// Graph norm + scatter + heads
// ---------------------------------------------------------------------------
__global__ void deg_init_kernel(float* deg) {
  const int n = blockIdx.x * blockDim.x + threadIdx.x;
  if (n < Nc) deg[n] = 1.0f;   // self-loop
}

__global__ void deg_count_kernel(const int* __restrict__ ei, float* deg) {
  const int e = blockIdx.x * blockDim.x + threadIdx.x;
  if (e < Ec) atomicAdd(&deg[ei[Ec + e]], 1.0f);   // segment_sum over dst
}

__global__ void dinv_kernel(const float* __restrict__ deg, float* __restrict__ dinv) {
  const int n = blockIdx.x * blockDim.x + threadIdx.x;
  if (n < Nc) dinv[n] = rsqrtf(deg[n]);            // deg >= 1 always
}

// agg init = self-loop term: xw[b,n,:] * dinv[n]^2  (also zero-initializes agg)
__global__ void self_kernel(const float* __restrict__ xw, const float* __restrict__ dinv,
                            float* __restrict__ agg) {
  const long idx = (long)blockIdx.x * blockDim.x + threadIdx.x;
  if (idx >= (long)BN * Hc) return;
  const int n = (int)((idx / Hc) % Nc);
  const float d = dinv[n];
  agg[idx] = xw[idx] * d * d;
}

// one wave per (batch, edge): 128 floats -> 4 per lane, f32 global atomics
__global__ __launch_bounds__(256) void scatter_kernel(
    const int* __restrict__ ei, const float* __restrict__ dinv,
    const float* __restrict__ xw, float* __restrict__ agg)
{
  const int lane = threadIdx.x & 31;
  const int widx = blockIdx.x * 8 + (threadIdx.x >> 5);
  if (widx >= Ec * Bc) return;
  const int e = widx % Ec;
  const int b = widx / Ec;
  const int s = ei[e];
  const int d = ei[Ec + e];
  const float nrm = dinv[s] * dinv[d];
  const float4 v = *(const float4*)&xw[((long)b * Nc + s) * Hc + lane * 4];
  float* dst = &agg[((long)b * Nc + d) * Hc + lane * 4];
  atomicAdd(dst + 0, v.x * nrm);
  atomicAdd(dst + 1, v.y * nrm);
  atomicAdd(dst + 2, v.z * nrm);
  atomicAdd(dst + 3, v.w * nrm);
}

// one wave per node: relu(agg + b) then two H->1 dots, wave32 shuffle reduce
__global__ __launch_bounds__(256) void head_kernel(
    const float* __restrict__ agg, const float* __restrict__ gcn_b,
    const float* __restrict__ wc, const float* __restrict__ bc,
    const float* __restrict__ wi, const float* __restrict__ bi,
    float* __restrict__ out)
{
  const int lane = threadIdx.x & 31;
  const int bn = blockIdx.x * 8 + (threadIdx.x >> 5);
  if (bn >= BN) return;
  float sc = 0.0f, si = 0.0f;
  const int j0 = lane * 4;
#pragma unroll
  for (int k = 0; k < 4; ++k) {
    const int j = j0 + k;
    float v = agg[(long)bn * Hc + j] + gcn_b[j];
    v = fmaxf(v, 0.0f);
    sc += v * wc[j];
    si += v * wi[j];
  }
#pragma unroll
  for (int off = 16; off > 0; off >>= 1) {
    sc += __shfl_xor(sc, off, 32);
    si += __shfl_xor(si, off, 32);
  }
  if (lane == 0) {
    out[bn]      = sc + bc[0];
    out[BN + bn] = si + bi[0];
  }
}

// ---------------------------------------------------------------------------
extern "C" void kernel_launch(void* const* d_in, const int* in_sizes, int n_in,
                              void* d_out, int out_size, void* d_ws, size_t ws_size,
                              hipStream_t stream)
{
  const float* dyn  = (const float*)d_in[0];
  const int*   ei   = (const int*)  d_in[1];
  const float* W_ih = (const float*)d_in[2];
  const float* W_hh = (const float*)d_in[3];
  const float* b_ih = (const float*)d_in[4];
  const float* b_hh = (const float*)d_in[5];
  const float* gcnW = (const float*)d_in[6];
  const float* gcnb = (const float*)d_in[7];
  const float* wc   = (const float*)d_in[8];
  const float* bc   = (const float*)d_in[9];
  const float* wi   = (const float*)d_in[10];
  const float* bi   = (const float*)d_in[11];
  float* out = (float*)d_out;

  float* ws   = (float*)d_ws;
  float* h    = ws;                         // BN*H
  float* xw   = h   + (long)BN * Hc;        // BN*H
  float* agg  = xw  + (long)BN * Hc;        // BN*H
  float* deg  = agg + (long)BN * Hc;        // N
  float* dinv = deg + Nc;                   // N

  deg_init_kernel <<<(Nc + 255) / 256, 256, 0, stream>>>(deg);
  deg_count_kernel<<<(Ec + 255) / 256, 256, 0, stream>>>(ei, deg);
  dinv_kernel     <<<(Nc + 255) / 256, 256, 0, stream>>>(deg, dinv);

  lstm_kernel<<<BN / 32, 256, 0, stream>>>(dyn, W_ih, W_hh, b_ih, b_hh, h);
  xw_kernel  <<<BN / 16, 256, 0, stream>>>(h, gcnW, xw);

  self_kernel   <<<(int)(((long)BN * Hc + 255) / 256), 256, 0, stream>>>(xw, dinv, agg);
  scatter_kernel<<<(Ec * Bc + 7) / 8, 256, 0, stream>>>(ei, dinv, xw, agg);
  head_kernel   <<<(BN + 7) / 8, 256, 0, stream>>>(agg, gcnb, wc, bc, wi, bi, out);
}
